// Net_71545565217443
// MI455X (gfx1250) — matrix operand order repaired and checked by reading, not compiled, not run on hardware
//
#include <hip/hip_runtime.h>
#include <hip/hip_bf16.h>

// Problem constants (match reference)
#define Bc 128
#define Tc 512
#define Ec 128
#define Hc 256
#define Mc 1024
#define Cc 5

typedef __attribute__((ext_vector_type(16))) _Float16 v16h;
typedef __attribute__((ext_vector_type(8)))  _Float16 v8h;
typedef __attribute__((ext_vector_type(8)))  float    v8f;

static __device__ __forceinline__ v8f wmma_f16(v16h a, v16h b, v8f c) {
  // (neg_a, A, neg_b, B, c_mod, C, reuse_a, reuse_b)
  return __builtin_amdgcn_wmma_f32_16x16x32_f16(false, a, false, b, (short)0, c,
                                                false, false);
}

// ---------------------------------------------------------------------------
// Kernel 1: Xproj[t*B+b][n] = emb[x[b,t]] @ W_ih^T + b_ih + b_hh   (f32 out)
// One 16x16 tile per wave, K=E=128 -> 4 wmma steps (fully unrolled).
// ---------------------------------------------------------------------------
__global__ __launch_bounds__(256) void xproj_kernel(
    const int* __restrict__ x, const float* __restrict__ emb,
    const float* __restrict__ W_ih, const float* __restrict__ b_ih,
    const float* __restrict__ b_hh, float* __restrict__ Xproj) {
  const int w     = threadIdx.x >> 5;
  const int lane  = threadIdx.x & 31;
  const int laneN = lane & 15;
  const int hi    = lane >> 4;
  const int tile  = blockIdx.x * 8 + w;   // 0..65535
  const int tm    = tile >> 4;            // 0..4095 (row tiles over B*T)
  const int tn    = tile & 15;            // 0..15   (col tiles over H)

  // A row for this lane: global row g = t*B + b
  const int g   = tm * 16 + laneN;
  const int t   = g >> 7;                 // /128
  const int b   = g & 127;
  const int tok = x[b * Tc + t];
  const float* arow = emb + (long)tok * Ec;
  const int n = tn * 16 + laneN;
  const float* brow = W_ih + (long)n * Ec;   // B[k][n] = W_ih[n][k]

  v8f c = {};
#pragma unroll
  for (int kk = 0; kk < Ec; kk += 32) {
    v16h a, bm;
    const int a0 = kk + hi * 8;
    const int a1 = kk + 16 + hi * 8;
#pragma unroll
    for (int j = 0; j < 8; ++j) {
      a[j]     = (_Float16)arow[a0 + j];
      a[8 + j] = (_Float16)arow[a1 + j];
    }
#pragma unroll
    for (int j = 0; j < 8; ++j) {
      bm[j]     = (_Float16)brow[a0 + j];
      bm[8 + j] = (_Float16)brow[a1 + j];
    }
    c = wmma_f16(a, bm, c);
  }
  const float bias = b_ih[n] + b_hh[n];
#pragma unroll
  for (int i = 0; i < 8; ++i) {
    const int gr = tm * 16 + i + hi * 8;       // C/D row map
    Xproj[(long)gr * Hc + n] = c[i] + bias;
  }
}

// ---------------------------------------------------------------------------
// Kernel 2: persistent single-WGP recurrence.
//   h <- active ? tanh(Xproj[t] + h @ W_hh^T) : 0
// (mask is monotone for right-aligned batching: inactive rows have never been
//  updated, so "keep old" == keep 0 -> no LDS read needed in write phase)
// W_hh f16 (128KB) + double-buffered f16 h (128KB) resident in LDS (320KB WGP).
// 32 waves x 4 tiles = 128 (8x16) output tiles; one barrier per step.
// Xproj accumulator-init loads for step t+1 are issued during step t
// (software pipeline) to hide L2 latency behind the WMMA chain + barrier.
// ---------------------------------------------------------------------------
__global__ __launch_bounds__(1024) void rnn_scan_kernel(
    const int* __restrict__ lengths, const float* __restrict__ W_hh,
    const float* __restrict__ Xproj, float* __restrict__ hn) {
  __shared__ __align__(64) _Float16 sW[Hc][Hc];     // 128 KB  sW[n][k]
  __shared__ __align__(64) _Float16 sH[2][Bc][Hc];  // 128 KB  double buffer

  const int tid = threadIdx.x;
  for (int i = tid; i < Hc * Hc; i += 1024)
    (&sW[0][0])[i] = (_Float16)W_hh[i];
  for (int i = tid; i < Bc * Hc; i += 1024)
    (&sH[0][0][0])[i] = (_Float16)0.0f;             // h0 = 0 in buffer 0
  __syncthreads();

  const int w = tid >> 5, lane = tid & 31;
  const int laneN = lane & 15, hi = lane >> 4;
  const int m   = w >> 2;          // 0..7  row-tile (fixed per wave)
  const int nt0 = (w & 3) * 4;     // col-tiles nt0..nt0+3
  const int rA  = m * 16 + laneN;  // A row (h row) this lane feeds

  int lenp[8];
#pragma unroll
  for (int i = 0; i < 8; ++i) lenp[i] = lengths[m * 16 + i + hi * 8];

  // Preload accumulator-init (Xproj tile) for t = 0.
  v8f acc[4];
#pragma unroll
  for (int j = 0; j < 4; ++j) {
    const int n = (nt0 + j) * 16 + laneN;
#pragma unroll
    for (int i = 0; i < 8; ++i)
      acc[j][i] = Xproj[(long)(m * 16 + i + hi * 8) * Hc + n];
  }

  for (int t = 0; t < Tc; ++t) {
    const int rb = t & 1, wb = rb ^ 1;

    // Issue next step's Xproj loads early (independent of this step's LDS).
    v8f accn[4];
    if (t + 1 < Tc) {
      const float* xp1 = Xproj + (long)(t + 1) * Bc * Hc;
#pragma unroll
      for (int j = 0; j < 4; ++j) {
        const int n = (nt0 + j) * 16 + laneN;
#pragma unroll
        for (int i = 0; i < 8; ++i)
          accn[j][i] = xp1[(long)(m * 16 + i + hi * 8) * Hc + n];
      }
    }

#pragma unroll
    for (int kk = 0; kk < Hc; kk += 32) {
      v16h a;
      const v8h lo  = *(const v8h*)&sH[rb][rA][kk + hi * 8];
      const v8h hiv = *(const v8h*)&sH[rb][rA][kk + 16 + hi * 8];
#pragma unroll
      for (int j = 0; j < 8; ++j) { a[j] = lo[j]; a[8 + j] = hiv[j]; }
#pragma unroll
      for (int j = 0; j < 4; ++j) {
        const int n = (nt0 + j) * 16 + laneN;
        const v8h b0v = *(const v8h*)&sW[n][kk + hi * 8];
        const v8h b1v = *(const v8h*)&sW[n][kk + 16 + hi * 8];
        v16h bm;
#pragma unroll
        for (int q = 0; q < 8; ++q) { bm[q] = b0v[q]; bm[8 + q] = b1v[q]; }
        acc[j] = wmma_f16(a, bm, acc[j]);
      }
    }

#pragma unroll
    for (int j = 0; j < 4; ++j) {
      const int n = (nt0 + j) * 16 + laneN;
#pragma unroll
      for (int i = 0; i < 8; ++i) {
        const int row = m * 16 + i + hi * 8;
        const bool act = (lenp[i] + t >= Tc);   // monotone activation
        sH[wb][row][n] = act ? (_Float16)tanhf(acc[j][i]) : (_Float16)0.0f;
      }
    }
    __syncthreads();

#pragma unroll
    for (int j = 0; j < 4; ++j) acc[j] = accn[j];
  }

  // T even -> final h lives in buffer 0
  for (int i = tid; i < Bc * Hc; i += 1024)
    hn[i] = (float)(&sH[0][0][0])[i];
}

// ---------------------------------------------------------------------------
// Kernel 3a: h1 = relu(hn @ W0^T + b0)   [128,1024], K=256. WMMA tiles.
// ---------------------------------------------------------------------------
__global__ __launch_bounds__(256) void mlp1_kernel(
    const float* __restrict__ hn, const float* __restrict__ W0,
    const float* __restrict__ b0, float* __restrict__ h1) {
  const int w = threadIdx.x >> 5, lane = threadIdx.x & 31;
  const int laneN = lane & 15, hi = lane >> 4;
  const int tile = blockIdx.x * 8 + w;   // 0..511
  const int tm = tile >> 6;              // 0..7
  const int tn = tile & 63;              // 0..63
  const int r = tm * 16 + laneN;
  const int n = tn * 16 + laneN;
  const float* arow = hn + (long)r * Hc;
  const float* brow = W0 + (long)n * Hc;

  v8f c = {};
#pragma unroll
  for (int kk = 0; kk < Hc; kk += 32) {
    v16h a, bm;
    const int a0 = kk + hi * 8, a1 = kk + 16 + hi * 8;
#pragma unroll
    for (int j = 0; j < 8; ++j) {
      a[j]     = (_Float16)arow[a0 + j];
      a[8 + j] = (_Float16)arow[a1 + j];
    }
#pragma unroll
    for (int j = 0; j < 8; ++j) {
      bm[j]     = (_Float16)brow[a0 + j];
      bm[8 + j] = (_Float16)brow[a1 + j];
    }
    c = wmma_f16(a, bm, c);
  }
  const float bias = b0[n];
#pragma unroll
  for (int i = 0; i < 8; ++i) {
    const int row = tm * 16 + i + hi * 8;
    const float v = c[i] + bias;
    h1[(long)row * Mc + n] = v > 0.f ? v : 0.f;
  }
}

// ---------------------------------------------------------------------------
// Kernel 3b: logits = relu(h1 @ W1^T + b1); out = log_softmax(logits)
// One block per batch row; K=1024, C=5 -> plain VALU reduction.
// ---------------------------------------------------------------------------
__global__ __launch_bounds__(64) void head_kernel(
    const float* __restrict__ h1, const float* __restrict__ W1,
    const float* __restrict__ b1, float* __restrict__ out) {
  __shared__ float red[Cc][64];
  __shared__ float logit[Cc];
  const int b = blockIdx.x, tid = threadIdx.x;
  float p[Cc] = {};
  for (int k = tid; k < Mc; k += 64) {
    const float hv = h1[(long)b * Mc + k];
#pragma unroll
    for (int c = 0; c < Cc; ++c) p[c] += hv * W1[c * Mc + k];
  }
#pragma unroll
  for (int c = 0; c < Cc; ++c) red[c][tid] = p[c];
  __syncthreads();
  if (tid < Cc) {
    float s = 0.f;
    for (int i = 0; i < 64; ++i) s += red[tid][i];
    s += b1[tid];
    logit[tid] = s > 0.f ? s : 0.f;
  }
  __syncthreads();
  if (tid == 0) {
    float mx = logit[0];
    for (int c = 1; c < Cc; ++c) mx = fmaxf(mx, logit[c]);
    float se = 0.f;
    for (int c = 0; c < Cc; ++c) se += expf(logit[c] - mx);
    const float lse = logf(se);
    for (int c = 0; c < Cc; ++c) out[b * Cc + c] = logit[c] - mx - lse;
  }
}

// ---------------------------------------------------------------------------
extern "C" void kernel_launch(void* const* d_in, const int* in_sizes, int n_in,
                              void* d_out, int out_size, void* d_ws,
                              size_t ws_size, hipStream_t stream) {
  const int*   x    = (const int*)d_in[0];
  const int*   len  = (const int*)d_in[1];
  const float* emb  = (const float*)d_in[2];
  const float* W_ih = (const float*)d_in[3];
  const float* W_hh = (const float*)d_in[4];
  const float* b_ih = (const float*)d_in[5];
  const float* b_hh = (const float*)d_in[6];
  const float* W0   = (const float*)d_in[7];
  const float* b0   = (const float*)d_in[8];
  const float* W1   = (const float*)d_in[9];
  const float* b1   = (const float*)d_in[10];
  float* out = (float*)d_out;

  // Workspace layout (f32): Xproj[T*B*H] | hn[B*H] | h1[B*M]  (~64.7 MB)
  float* Xproj = (float*)d_ws;
  float* hn    = Xproj + (long)Tc * Bc * Hc;
  float* h1    = hn + (long)Bc * Hc;

  // 65536 tiles / 8 waves-per-block
  xproj_kernel<<<8192, 256, 0, stream>>>(x, emb, W_ih, b_ih, b_hh, Xproj);
  rnn_scan_kernel<<<1, 1024, 0, stream>>>(len, W_hh, Xproj, hn);
  mlp1_kernel<<<64, 256, 0, stream>>>(hn, W0, b0, h1);
  head_kernel<<<Bc, 64, 0, stream>>>(h1, W1, b1, out);
}
